// GraphAttentionLayer_83837761618207
// MI455X (gfx1250) — matrix operand order repaired and checked by reading, not compiled, not run on hardware
//
#include <hip/hip_runtime.h>
#include <hip/hip_bf16.h>

typedef __attribute__((ext_vector_type(16))) _Float16 v16h;
typedef __attribute__((ext_vector_type(8)))  _Float16 v8h;
typedef __attribute__((ext_vector_type(8)))  float    v8f;
typedef __attribute__((ext_vector_type(4)))  int      v4i;

#define ALPHA    0.2f
#define NEG_INF  (-9.0e15f)
#define N        2048
#define F        128
#define B        8
#define LROW     40          // padded LDS row stride in halves (80B: bank-spread, 16B aligned)

#if defined(__has_builtin)
#  if __has_builtin(__builtin_amdgcn_global_load_async_to_lds_b128) && \
      __has_builtin(__builtin_amdgcn_s_wait_asynccnt)
#    define HAVE_ASYNC_LDS 1
#  endif
#endif
#ifndef HAVE_ASYNC_LDS
#  define HAVE_ASYNC_LDS 0
#endif

#if HAVE_ASYNC_LDS
typedef __attribute__((address_space(1))) v4i gv4i;   // global-AS 16B chunk
typedef __attribute__((address_space(3))) v4i lv4i;   // LDS-AS   16B chunk
#endif

// ---- fragment loaders ------------------------------------------------------
// B-operand (16-bit, 32x16): lane n (n<16) holds col N=n, K=k0..k0+7 and k0+16..k0+23;
// lane n+16 holds K=k0+8..k0+15 and k0+24..k0+31.  Two 16B runs.
__device__ __forceinline__ v16h load_fragB_h(const _Float16* p) {
    v8h lo = *(const v8h*)p;
    v8h hi = *(const v8h*)(p + 16);
    return __builtin_shufflevector(lo, hi, 0,1,2,3,4,5,6,7,8,9,10,11,12,13,14,15);
}

// ---- kernel 1: Wt (f16 transpose of W) + wa1/wa2 = W @ a1/a2 ---------------
__global__ void gat_prep_kernel(const float* __restrict__ W,
                                const float* __restrict__ a,
                                _Float16* __restrict__ Wt,
                                float* __restrict__ wa) {
    int k = threadIdx.x;                 // 0..127 : row of W
    float s1 = 0.f, s2 = 0.f;
    for (int f = 0; f < F; ++f) {
        float w = W[k * F + f];
        s1 += w * a[f];
        s2 += w * a[F + f];
        Wt[f * F + k] = (_Float16)w;     // Wt[f][k] = W[k][f]
    }
    wa[k]     = s1;
    wa[F + k] = s2;
}

// ---- kernel 2: masked distance bias (computed once, reused across batch) ---
__global__ void gat_bias_kernel(const int4* __restrict__ adj4,
                                const float4* __restrict__ dist4,
                                const float* __restrict__ dsc,
                                float4* __restrict__ mb4) {
    float s = dsc[0];
    int total4 = (N * N) / 4;
    for (int i = blockIdx.x * blockDim.x + threadIdx.x; i < total4;
         i += gridDim.x * blockDim.x) {
        int4   a = adj4[i];
        float4 d = dist4[i];
        float4 r;
        r.x = (a.x > 0) ? -__logf(d.x + 1e-6f) * s : NEG_INF;
        r.y = (a.y > 0) ? -__logf(d.y + 1e-6f) * s : NEG_INF;
        r.z = (a.z > 0) ? -__logf(d.z + 1e-6f) * s : NEG_INF;
        r.w = (a.w > 0) ? -__logf(d.w + 1e-6f) * s : NEG_INF;
        mb4[i] = r;
    }
}

// ---- kernel 3: h_prime = h @ W, stored transposed as f16 hpT[b][f][n] ------
__global__ void gat_hprime_kernel(const float* __restrict__ h,
                                  const _Float16* __restrict__ Wt,
                                  _Float16* __restrict__ hpT) {
    int lane = threadIdx.x & 31, wid = threadIdx.x >> 5;
    int strip = blockIdx.x * 8 + wid;     // 0..1023, 16 rows each
    int g0 = strip * 16;                  // global row (b*N+n)
    int b  = g0 >> 11;
    int n0 = g0 & (N - 1);
    int row  = lane & 15;
    int hi8  = (lane >> 4) << 3;
    int hi16 = (lane >> 4) << 4;

    // A fragments: 16 rows x 128 K (4 fragments of K=32), f32 -> f16
    const float* hrow = h + (long)(g0 + row) * F;
    v16h afr[4];
    #pragma unroll
    for (int q = 0; q < 4; ++q) {
        int kb = q * 32 + hi8;
        float lo[8], hi[8];
        *(float4*)(lo)     = *(const float4*)(hrow + kb);
        *(float4*)(lo + 4) = *(const float4*)(hrow + kb + 4);
        *(float4*)(hi)     = *(const float4*)(hrow + kb + 16);
        *(float4*)(hi + 4) = *(const float4*)(hrow + kb + 20);
        v16h r;
        #pragma unroll
        for (int t = 0; t < 8; ++t) { r[t] = (_Float16)lo[t]; r[t + 8] = (_Float16)hi[t]; }
        afr[q] = r;
    }

    _Float16* hb = hpT + (long)b * F * N;
    #pragma unroll
    for (int c = 0; c < 8; ++c) {
        v8f acc = {0.f,0.f,0.f,0.f,0.f,0.f,0.f,0.f};
        #pragma unroll
        for (int q = 0; q < 4; ++q) {
            v16h bf = load_fragB_h(Wt + (c * 16 + row) * F + q * 32 + hi8);
            acc = __builtin_amdgcn_wmma_f32_16x16x32_f16(
                false, afr[q], false, bf, (short)0, acc, false, false);
        }
        // C layout: VGPR r -> row n0+r+hi8, col f = c*16+row.
        v8h st;
        #pragma unroll
        for (int r = 0; r < 8; ++r) st[r] = (_Float16)acc[r];
        *(v8h*)(hb + (long)(c * 16 + row) * N + n0 + hi8) = st;
    }
}

// ---- kernel 4: si = h@wa1, sj = h@wa2 (one wave per row) -------------------
__global__ void gat_sisj_kernel(const float* __restrict__ h,
                                const float* __restrict__ wa,
                                float* __restrict__ si,
                                float* __restrict__ sj) {
    int lane = threadIdx.x & 31, wid = threadIdx.x >> 5;
    int g = blockIdx.x * 8 + wid;         // 0..16383
    const float* hr = h + (long)g * F;
    float p1 = 0.f, p2 = 0.f;
    #pragma unroll
    for (int s = 0; s < 4; ++s) {
        float v = hr[lane + s * 32];
        p1 += v * wa[lane + s * 32];
        p2 += v * wa[F + lane + s * 32];
    }
    #pragma unroll
    for (int d = 16; d >= 1; d >>= 1) {
        p1 += __shfl_xor(p1, d, 32);
        p2 += __shfl_xor(p2, d, 32);
    }
    if (lane == 0) { si[g] = p1; sj[g] = p2; }
}

// ---- cooperative V-tile stager: hpT[b][f][j0..j0+31] -> LDS [f][0..31] -----
// 128 threads: thread f stages one 64B row (4 x b128).
__device__ __forceinline__ void stage_tile(const _Float16* __restrict__ hb,
                                           int tid, int j0,
                                           _Float16* __restrict__ dst) {
    const _Float16* g = hb + (long)tid * N + j0;
    _Float16* l = dst + tid * LROW;
#if HAVE_ASYNC_LDS
    gv4i* gp = (gv4i*)(void*)const_cast<_Float16*>(g);
    lv4i* lp = (lv4i*)(void*)l;
    __builtin_amdgcn_global_load_async_to_lds_b128(gp + 0, lp + 0, 0, 0);
    __builtin_amdgcn_global_load_async_to_lds_b128(gp + 1, lp + 1, 0, 0);
    __builtin_amdgcn_global_load_async_to_lds_b128(gp + 2, lp + 2, 0, 0);
    __builtin_amdgcn_global_load_async_to_lds_b128(gp + 3, lp + 3, 0, 0);
#else
    v8h r0 = *(const v8h*)(g);
    v8h r1 = *(const v8h*)(g + 8);
    v8h r2 = *(const v8h*)(g + 16);
    v8h r3 = *(const v8h*)(g + 24);
    *(v8h*)(l)      = r0;
    *(v8h*)(l + 8)  = r1;
    *(v8h*)(l + 16) = r2;
    *(v8h*)(l + 24) = r3;
#endif
}

// ---- kernel 5: fused online-softmax attention + P@V + ELU ------------------
// 4 waves/block share batch + j-sweep -> V tile staged in LDS once per block.
__global__ void __launch_bounds__(128)
gat_attn_kernel(const _Float16* __restrict__ hpT,
                const float* __restrict__ mbias,
                const float* __restrict__ si_,
                const float* __restrict__ sj_,
                float* __restrict__ out) {
    __shared__ _Float16 vbuf[2][F * LROW];   // 2 x 10KB double buffer

    int tid  = threadIdx.x;
    int lane = tid & 31, wid = tid >> 5;
    int task = blockIdx.x * 4 + wid;      // 0..1023
    int b  = task >> 7;                   // all 4 waves: same batch
    int i0 = (task & 127) * 16;
    int row  = lane & 15;
    int hi8  = (lane >> 4) << 3;
    int hi16 = (lane >> 4) << 4;

    const _Float16* hb   = hpT + (long)b * F * N;
    const float*    sjb  = sj_ + b * N;
    const float*    mrow = mbias + (long)(i0 + row) * N;
    float si_i = si_[b * N + i0 + row];

    float m_run = -INFINITY, l_run = 0.f;
    v8f acc[8];
    #pragma unroll
    for (int c = 0; c < 8; ++c) acc[c] = (v8f){0.f,0.f,0.f,0.f,0.f,0.f,0.f,0.f};

    stage_tile(hb, tid, 0, &vbuf[0][0]);   // prologue: tile 0 -> buf 0

    for (int jt = 0; jt < N / 32; ++jt) {
        int j0  = jt * 32;
        int cur = jt & 1;
#if HAVE_ASYNC_LDS
        __builtin_amdgcn_s_wait_asynccnt(0);
#endif
        __syncthreads();                   // buf[cur] ready; buf[cur^1] free
        if (jt + 1 < N / 32)
            stage_tile(hb, tid, j0 + 32, &vbuf[cur ^ 1][0]);

        // ---- scores for 16x32 tile (A-operand K mapping) ----
        int jb = j0 + hi8;
        float sjv[16], mbv[16];
        *(float4*)(sjv)      = *(const float4*)(sjb + jb);
        *(float4*)(sjv + 4)  = *(const float4*)(sjb + jb + 4);
        *(float4*)(sjv + 8)  = *(const float4*)(sjb + jb + 16);
        *(float4*)(sjv + 12) = *(const float4*)(sjb + jb + 20);
        *(float4*)(mbv)      = *(const float4*)(mrow + jb);
        *(float4*)(mbv + 4)  = *(const float4*)(mrow + jb + 4);
        *(float4*)(mbv + 8)  = *(const float4*)(mrow + jb + 16);
        *(float4*)(mbv + 12) = *(const float4*)(mrow + jb + 20);

        if (jt + 1 < N / 32) __builtin_prefetch(mrow + jb + 32, 0, 1);

        float sc[16];
        float mloc = -INFINITY;
        #pragma unroll
        for (int t = 0; t < 16; ++t) {
            float x = si_i + sjv[t];
            x = (x > 0.f) ? x : ALPHA * x;   // leaky relu
            x += mbv[t];                     // masked distance bias
            sc[t] = x;
            mloc = fmaxf(mloc, x);
        }
        float mtile = fmaxf(mloc, __shfl_xor(mloc, 16, 32));
        float mnew  = fmaxf(m_run, mtile);
        float ratio = __expf(m_run - mnew);  // 0 on first tile (m_run = -inf)

        float psum = 0.f;
        v16h pf;
        #pragma unroll
        for (int t = 0; t < 16; ++t) {
            float p = __expf(sc[t] - mnew);
            psum += p;
            pf[t] = (_Float16)p;
        }
        psum += __shfl_xor(psum, 16, 32);
        l_run = l_run * ratio + psum;
        m_run = mnew;

        // rescale only when some row's max actually moved (rare after warmup)
        if (__any(ratio != 1.0f)) {
            #pragma unroll
            for (int r = 0; r < 8; ++r) {
                float s = __shfl(ratio, r + hi8, 32);
                #pragma unroll
                for (int c = 0; c < 8; ++c) acc[c][r] *= s;
            }
        }

        // ---- P (16x32) @ V (32x128): 8 ds-fed fragments, then 8 WMMAs ----
        const _Float16* vb = &vbuf[cur][0];
        v16h vf[8];
        #pragma unroll
        for (int c = 0; c < 8; ++c) {
            const _Float16* p = vb + (c * 16 + row) * LROW + hi8;
            v8h lo = *(const v8h*)p;
            v8h hi = *(const v8h*)(p + 16);
            vf[c] = __builtin_shufflevector(lo, hi,
                        0,1,2,3,4,5,6,7,8,9,10,11,12,13,14,15);
        }
        #pragma unroll
        for (int c = 0; c < 8; ++c)
            acc[c] = __builtin_amdgcn_wmma_f32_16x16x32_f16(
                false, pf, false, vf[c], (short)0, acc[c], false, false);
    }

    // normalize, ELU, store
    float linv = 1.f / l_run;
    float* ob = out + ((long)b * N + i0) * F;
    #pragma unroll
    for (int r = 0; r < 8; ++r) {
        float lr = __shfl(linv, r + hi8, 32);
        #pragma unroll
        for (int c = 0; c < 8; ++c) {
            float x = acc[c][r] * lr;
            x = (x > 0.f) ? x : (__expf(x) - 1.f);
            ob[(long)(r + hi8) * F + c * 16 + row] = x;
        }
    }
}

// ---- launcher --------------------------------------------------------------
extern "C" void kernel_launch(void* const* d_in, const int* in_sizes, int n_in,
                              void* d_out, int out_size, void* d_ws, size_t ws_size,
                              hipStream_t stream) {
    const float* h    = (const float*)d_in[0];
    const int*   adj  = (const int*)  d_in[1];
    const float* dist = (const float*)d_in[2];
    const float* W    = (const float*)d_in[3];
    const float* a    = (const float*)d_in[4];
    const float* dsc  = (const float*)d_in[5];
    float* out = (float*)d_out;

    char* ws = (char*)d_ws;
    float*    mbias = (float*)(ws);                       // 2048*2048*4  = 16 MiB
    _Float16* hpT   = (_Float16*)(ws + 16777216);         // 8*128*2048*2 =  4 MiB
    float*    si    = (float*)(ws + 20971520);            // 8*2048*4
    float*    sj    = (float*)(ws + 21037056);            // 8*2048*4
    _Float16* Wt    = (_Float16*)(ws + 21102592);         // 128*128*2
    float*    wa    = (float*)(ws + 21135360);            // 256*4

    gat_prep_kernel  <<<1,    128, 0, stream>>>(W, a, Wt, wa);
    gat_bias_kernel  <<<1024, 256, 0, stream>>>((const int4*)adj, (const float4*)dist,
                                                dsc, (float4*)mbias);
    gat_hprime_kernel<<<128,  256, 0, stream>>>(h, Wt, hpT);
    gat_sisj_kernel  <<<2048, 256, 0, stream>>>(h, wa, si, sj);
    gat_attn_kernel  <<<256,  128, 0, stream>>>(hpT, mbias, si, sj, out);
}
